// Point_contextual_entropy_parameter_6047313953087
// MI455X (gfx1250) — matrix operand order repaired and checked by reading, not compiled, not run on hardware
//
#include <hip/hip_runtime.h>
#include <stdint.h>

typedef __attribute__((ext_vector_type(16))) __bf16   v16bf;
typedef __attribute__((ext_vector_type(8)))  float    v8f;
typedef __attribute__((ext_vector_type(4)))  uint32_t u32x4;

struct U32x8 { u32x4 lo, hi; };   // 32 bytes == one v16bf fragment

#define ZROW 65536                // zero-filled row appended to every feature buffer

__device__ __forceinline__ uint16_t f32_to_bf16(float f) {
  uint32_t u = __builtin_bit_cast(uint32_t, f);
  u += 0x7fffu + ((u >> 16) & 1u);          // round-to-nearest-even
  return (uint16_t)(u >> 16);
}

// A-fragment: lane's K-runs are 16 elements apart -> loads at p and p+16 (ushorts)
__device__ __forceinline__ v16bf load_frag_a(const uint16_t* p) {
  U32x8 d;
  d.lo = *(const u32x4*)p;
  d.hi = *(const u32x4*)(p + 16);
  return __builtin_bit_cast(v16bf, d);
}

// ---- elementwise f32 -> bf16 (2 elements / thread, packed dword stores) ----
__global__ void convert_f32_to_bf16(const float* __restrict__ src,
                                    uint16_t* __restrict__ dst, int n2) {
  int i = blockIdx.x * blockDim.x + threadIdx.x;
  if (i < n2) {
    float a = src[2 * i], b = src[2 * i + 1];
    uint32_t packed = (uint32_t)f32_to_bf16(a) | ((uint32_t)f32_to_bf16(b) << 16);
    ((uint32_t*)dst)[i] = packed;
  }
}

// ---- zero the padding row (index ZROW) of the three feature buffers ----
__global__ void zero_pad_rows(uint32_t* xbf, uint32_t* h1, uint32_t* h2) {
  int t = threadIdx.x;                       // 96 threads, 32 dwords per row
  const size_t off = (size_t)ZROW * 32;
  if (t < 32)       xbf[off + t]        = 0u;
  else if (t < 64)  h1[off + (t - 32)]  = 0u;
  else              h2[off + (t - 64)]  = 0u;
}

// ---- pack W[k][cin][cout] (f32) into WMMA B-fragment-major bf16 layout ----
// frag = (k*2 + s)*T + t (512 bf16); within frag: lane*16 + (j*2+p)
// value = W[k][s*32 + (j>>2)*16 + hs*8 + 2*(j&3)+p][t*16 + m]   (B layout per ISA 7.12.2)
__global__ void pack_weights(const float* __restrict__ W,
                             uint16_t* __restrict__ Wp, int cout) {
  int idx = blockIdx.x * blockDim.x + threadIdx.x;
  int T = cout >> 4;
  int total = 27 * 2 * T * 512;
  if (idx >= total) return;
  int e    = idx & 511;
  int frag = idx >> 9;
  int t  = frag % T;
  int ks = frag / T;
  int s = ks & 1, k = ks >> 1;
  int lane = e >> 4;
  int j = (e >> 1) & 7;
  int p = e & 1;
  int m = lane & 15, hs = lane >> 4;
  int kk = s * 32 + (j >> 2) * 16 + hs * 8 + 2 * (j & 3) + p;
  int n  = t * 16 + m;
  Wp[idx] = f32_to_bf16(W[(k * 64 + kk) * cout + n]);
}

// ---- sparse-conv layer: 8 waves/block, one 32-row x COUT tile per wave -----
// B (weights) staged through LDS once per BLOCK (8x less L2 traffic),
// double-buffered: global prefetch of slice k+1 overlaps the 16-WMMA burst
// on slice k; one __syncthreads per iteration. A stays a direct L2 gather
// with zero-row clamp and one-ahead index prefetch.
template <int COUT, bool RELU, bool OUT_F32>
__global__ __launch_bounds__(256)
void spconv_layer(const uint16_t* __restrict__ feat,   // [ZROW+1, 64] bf16
                  const int* __restrict__ nidx,        // [N, 27]
                  const uint16_t* __restrict__ Wp,     // packed bf16
                  const float* __restrict__ bias,      // [COUT] f32
                  void* __restrict__ outv) {
  constexpr int T = COUT / 16;
  constexpr int BWORDS = 2 * T * 512;        // ushorts per k-slice (8KB for T=4)
  __shared__ uint16_t sB[2][BWORDS];

  const int tid     = threadIdx.x;
  const int lane    = tid & 31;
  const int wave    = (blockIdx.x * blockDim.x + tid) >> 5;
  const int rowbase = wave * 32;
  const int m  = lane & 15;
  const int hs = lane >> 4;
  const int ib0 = (rowbase + m) * 27;        // nidx base, M-tile 0
  const int ib1 = (rowbase + 16 + m) * 27;   // nidx base, M-tile 1

  v8f acc[2][T];
#pragma unroll
  for (int mt = 0; mt < 2; ++mt)
#pragma unroll
    for (int t = 0; t < T; ++t)
      acc[mt][t] = (v8f){0.f, 0.f, 0.f, 0.f, 0.f, 0.f, 0.f, 0.f};

  // stage B[0] cooperatively: 256 threads x (8T bytes each)
  {
    const uint2* wsrc = (const uint2*)(Wp) + tid * T;
    uint2*       sdst = (uint2*)(&sB[0][0]) + tid * T;
#pragma unroll
    for (int i = 0; i < T; ++i) sdst[i] = wsrc[i];
  }
  __syncthreads();

  // prefetch idx[0]
  int rn0 = nidx[ib0];
  int rn1 = nidx[ib1];
  int buf = 0;

  for (int k = 0; k < 27; ++k) {
    // global prefetch of B slice k+1 into registers (clamped; tail redundant)
    int kp = (k + 1 < 27) ? (k + 1) : 26;
    const uint2* wsrc = (const uint2*)(Wp + (size_t)kp * BWORDS) + tid * T;
    uint2 pre[T];
#pragma unroll
    for (int i = 0; i < T; ++i) pre[i] = wsrc[i];

    // A[k]: rows from prefetched indices, zero-row clamp, unconditional loads
    int c0 = rn0 < 0 ? ZROW : rn0;
    int c1 = rn1 < 0 ? ZROW : rn1;
    const uint16_t* p0 = feat + (size_t)c0 * 64 + hs * 8;
    const uint16_t* p1 = feat + (size_t)c1 * 64 + hs * 8;
    v16bf a00 = load_frag_a(p0);
    v16bf a01 = load_frag_a(p0 + 32);
    v16bf a10 = load_frag_a(p1);
    v16bf a11 = load_frag_a(p1 + 32);

    // idx[k+1] prefetch (clamped; dead value on last iteration)
    int k1 = (k + 1 < 27) ? k + 1 : 0;
    rn0 = nidx[ib0 + k1];
    rn1 = nidx[ib1 + k1];

    // B[k] fragments from LDS: contiguous 32B per lane per fragment
    const uint16_t* sb = &sB[buf][0] + lane * 16;
    U32x8 b[2][T];
#pragma unroll
    for (int s = 0; s < 2; ++s)
#pragma unroll
      for (int t = 0; t < T; ++t) {
        b[s][t].lo = *(const u32x4*)(sb + (s * T + t) * 512);
        b[s][t].hi = *(const u32x4*)(sb + (s * T + t) * 512 + 8);
      }

    // 16 back-to-back WMMAs on A[k] x B[k]
#pragma unroll
    for (int s = 0; s < 2; ++s) {
      v16bf a0s = (s == 0) ? a00 : a01;
      v16bf a1s = (s == 0) ? a10 : a11;
#pragma unroll
      for (int t = 0; t < T; ++t) {
        v16bf bf = __builtin_bit_cast(v16bf, b[s][t]);
        acc[0][t] = __builtin_amdgcn_wmma_f32_16x16x32_bf16(
            false, a0s, false, bf, (short)0, acc[0][t], false, false);
        acc[1][t] = __builtin_amdgcn_wmma_f32_16x16x32_bf16(
            false, a1s, false, bf, (short)0, acc[1][t], false, false);
      }
    }

    // commit prefetched slice to the other buffer; safe: that buffer's reads
    // finished before the PREVIOUS barrier. One barrier per iteration.
    {
      uint2* sdst = (uint2*)(&sB[buf ^ 1][0]) + tid * T;
#pragma unroll
      for (int i = 0; i < T; ++i) sdst[i] = pre[i];
    }
    __syncthreads();
    buf ^= 1;
  }

  // epilogue: C layout -> VGPR r holds rows (hs?8:0)+r, column = t*16 + m
#pragma unroll
  for (int mt = 0; mt < 2; ++mt) {
#pragma unroll
    for (int t = 0; t < T; ++t) {
      float bv = bias[t * 16 + m];
#pragma unroll
      for (int r = 0; r < 8; ++r) {
        float v = acc[mt][t][r] + bv;
        if (RELU) v = v > 0.f ? v : 0.f;
        int row = rowbase + mt * 16 + hs * 8 + r;
        int col = t * 16 + m;
        if (OUT_F32) ((float*)outv)[row * COUT + col] = v;
        else         ((uint16_t*)outv)[row * COUT + col] = f32_to_bf16(v);
      }
    }
  }
}

extern "C" void kernel_launch(void* const* d_in, const int* in_sizes, int n_in,
                              void* d_out, int out_size, void* d_ws, size_t ws_size,
                              hipStream_t stream) {
  const float* x  = (const float*)d_in[0];
  const int*  nid = (const int*)d_in[1];
  const float* W1 = (const float*)d_in[2];
  const float* b1 = (const float*)d_in[3];
  const float* W2 = (const float*)d_in[4];
  const float* b2 = (const float*)d_in[5];
  const float* W3 = (const float*)d_in[6];
  const float* b3 = (const float*)d_in[7];

  constexpr int N = 65536;

  // workspace (bytes): xbf | h1 | h2 (each (N+1) rows x 64 bf16, 8.5MiB slots)
  char* ws = (char*)d_ws;
  uint16_t* xbf = (uint16_t*)(ws);
  uint16_t* h1  = (uint16_t*)(ws + (size_t)8912896);
  uint16_t* h2  = (uint16_t*)(ws + (size_t)17825792);
  uint16_t* W1p = (uint16_t*)(ws + (size_t)26738688);
  uint16_t* W2p = (uint16_t*)(ws + (size_t)26738688 + 262144);
  uint16_t* W3p = (uint16_t*)(ws + (size_t)26738688 + 524288);

  int n2 = N * 64 / 2;
  convert_f32_to_bf16<<<(n2 + 255) / 256, 256, 0, stream>>>(x, xbf, n2);
  zero_pad_rows<<<1, 96, 0, stream>>>((uint32_t*)xbf, (uint32_t*)h1, (uint32_t*)h2);
  pack_weights<<<(27 * 64 * 64 + 255) / 256, 256, 0, stream>>>(W1, W1p, 64);
  pack_weights<<<(27 * 64 * 64 + 255) / 256, 256, 0, stream>>>(W2, W2p, 64);
  pack_weights<<<(27 * 64 * 16 + 255) / 256, 256, 0, stream>>>(W3, W3p, 16);

  int blocks = (N / 32) / 8;  // 8 waves (32-row tiles) per 256-thread block
  spconv_layer<64, true,  false><<<blocks, 256, 0, stream>>>(xbf, nid, W1p, b1, h1);
  spconv_layer<64, true,  false><<<blocks, 256, 0, stream>>>(h1,  nid, W2p, b2, h2);
  spconv_layer<16, false, true ><<<blocks, 256, 0, stream>>>(h2,  nid, W3p, b3, d_out);
}
